// ExpertSelectiveTimeVaryingSSM_15934328668593
// MI455X (gfx1250) — matrix operand order repaired and checked by reading, not compiled, not run on hardware
//
#include <hip/hip_runtime.h>
#include <hip/hip_bf16.h>
#include <math.h>

// ---------------------------------------------------------------------------
// Types for CDNA5 WMMA (wave32): D(16x16 f32) = A(16x32 f16) x B(32x16 f16) + C
// ---------------------------------------------------------------------------
typedef __attribute__((ext_vector_type(16))) _Float16 v16h;
typedef __attribute__((ext_vector_type(8)))  _Float16 v8h;
typedef __attribute__((ext_vector_type(8)))  float    v8f;
typedef __attribute__((ext_vector_type(4)))  unsigned int u32x4;
typedef __attribute__((ext_vector_type(8)))  int      i32x8;
typedef __attribute__((ext_vector_type(4)))  int      i32x4;

#define BB 64
#define TT 1024
#define DS 128
#define DU 64
#define DY 64
#define MM 8
#define GHH 64

#if defined(__has_builtin)
#if __has_builtin(__builtin_amdgcn_tensor_load_to_lds) && __has_builtin(__builtin_amdgcn_s_wait_tensorcnt)
#define HAVE_TDM 1
#endif
#endif
#ifndef HAVE_TDM
#define HAVE_TDM 0
#endif

__device__ __forceinline__ v8f wmma16(v16h a, v16h b, v8f c) {
  return __builtin_amdgcn_wmma_f32_16x16x32_f16(
      /*neg_a=*/false, a, /*neg_b=*/false, b,
      /*c_mod=*/(short)0, c, /*reuse_a=*/false, /*reuse_b=*/false);
}

// 16x32 f16 A-fragment straight from an f16 LDS row (contiguous K layout):
// lane L (M=L&15) holds K = 2*kb..2*kb+15 (elems 0..15),
// kb = (L&16)?8:0 plus the caller's chunk offset. Two aligned 16B loads.
__device__ __forceinline__ v16h a_frag_raw(const _Float16* src, int kb) {
  v8h lo = *(const v8h*)(src + 2 * kb);
  v8h hi = *(const v8h*)(src + 2 * kb + 8);
  v16h a;
#pragma unroll
  for (int e = 0; e < 8; ++e) { a[e] = lo[e]; a[8 + e] = hi[e]; }
  return a;
}

__device__ __forceinline__ v16h frag_scale(v16h z, _Float16 s) {
  v16h a;
#pragma unroll
  for (int e = 0; e < 16; ++e) a[e] = z[e] * s;   // packs into v_pk_mul_f16
  return a;
}

// ---------------------------------------------------------------------------
// TDM: 2D tile (f16) Global -> LDS. D# per ISA cdna5 ch.8:
//  group0: count=1 | lds_addr[63:32] | global_addr[120:64] | type=2 [127:126]
//  group1: data_size=1(2B) [17:16]; tensor_dim0/1; tile_dim0/1; dim strides.
// All extents in elements (data_size units). Groups 2/3 (and the trailing
// int32x8) are only used for >2D tensors -> zero for our 2D tiles.
// Issued by one wave; completion via s_wait_tensorcnt 0 + workgroup barrier.
// ---------------------------------------------------------------------------
#if HAVE_TDM
__device__ __forceinline__ void tdm_load_2d_f16(void* lds_dst, const void* gsrc,
                                                unsigned tile_w, unsigned tile_h,
                                                unsigned tensor_w, unsigned tensor_h,
                                                unsigned row_stride) {
  unsigned lds_addr = (unsigned)(unsigned long long)(uintptr_t)lds_dst; // LDS byte offset
  unsigned long long ga = (unsigned long long)(uintptr_t)gsrc;
  u32x4 g0;
  g0[0] = 1u;                                         // count=1 (valid user D#)
  g0[1] = lds_addr;                                   // lds_addr
  g0[2] = (unsigned)(ga & 0xffffffffu);               // global_addr[31:0]
  g0[3] = (unsigned)((ga >> 32) & 0x01ffffffu) | (2u << 30);  // addr[56:32] | type=2
  unsigned long long plane = (unsigned long long)tensor_w * tensor_h;
  i32x8 g1;
  g1[0] = (int)(1u << 16);                            // data_size = 1 -> 2 bytes
  g1[1] = (int)((tensor_w & 0xffffu) << 16);          // tensor_dim0[15:0] @ bits63:48
  g1[2] = (int)(((tensor_w >> 16) & 0xffffu) | ((tensor_h & 0xffffu) << 16));
  g1[3] = (int)(((tensor_h >> 16) & 0xffffu) | ((tile_w & 0xffffu) << 16));
  g1[4] = (int)(tile_h & 0xffffu);                    // tile_dim1; tile_dim2=0
  g1[5] = (int)row_stride;                            // tensor_dim0_stride[31:0]
  g1[6] = (int)(((unsigned)(plane & 0xffffu)) << 16); // stride[47:32]=0 | dim1_stride[15:0]
  g1[7] = (int)((plane >> 16) & 0xffffffffu);         // dim1_stride[47:16]
  i32x4 z4; z4[0] = 0; z4[1] = 0; z4[2] = 0; z4[3] = 0;
  i32x8 z8;
#pragma unroll
  for (int i = 0; i < 8; ++i) z8[i] = 0;
  __builtin_amdgcn_tensor_load_to_lds(g0, g1, z4, z4, z8, 0);
}
#endif

// ---------------------------------------------------------------------------
// Kernel 0a: S = tril(S_raw), diag -> softplus(diag)+eps; Sinv by fwd subst.
// One block, 128 threads (thread c solves column c). Also emits f16 Sinv.
// ---------------------------------------------------------------------------
__global__ __launch_bounds__(128) void k_buildS(const float* __restrict__ S_raw,
                                                float* __restrict__ Sout,
                                                float* __restrict__ Sinv,
                                                _Float16* __restrict__ Sinv_h) {
  __shared__ float Ss[DS][DS];
  int c = threadIdx.x;
  for (int i = 0; i < DS; ++i) {
    float vv = S_raw[i * DS + c];
    float sv = (c > i) ? 0.f : vv;
    if (c == i) {
      float sp = (vv > 20.f) ? vv : log1pf(expf(vv));
      sv = sp + 0.001f;
    }
    Ss[i][c] = sv;
    Sout[i * DS + c] = sv;
  }
  __syncthreads();
  float x[DS];
  for (int i = 0; i < DS; ++i) {
    float a = (i == c) ? 1.0f : 0.0f;
    for (int k = 0; k < i; ++k) a -= Ss[i][k] * x[k];
    x[i] = a / Ss[i][i];
  }
  for (int i = 0; i < DS; ++i) {
    Sinv[i * DS + c] = x[i];
    Sinv_h[i * DS + c] = (_Float16)x[i];
  }
}

// ---------------------------------------------------------------------------
// Kernel 0b: contractive experts. Zero block [128:,128:], spectral norm via
// power iteration on A^T A, scale by 1/max(sigma,1). One block per expert.
// ---------------------------------------------------------------------------
__global__ __launch_bounds__(256) void k_experts(const float* __restrict__ K_raw,
                                                 float* __restrict__ Kexp) {
  __shared__ float A[192][192];
  __shared__ float v[192];
  __shared__ float w[192];
  __shared__ float snorm;
  int m = blockIdx.x, tid = threadIdx.x;
  for (int idx = tid; idx < 192 * 192; idx += 256) {
    int i = idx / 192, j = idx - i * 192;
    float val = K_raw[(m * 192 + i) * 192 + j];
    if (i >= DS && j >= DS) val = 0.f;
    A[i][j] = val;
  }
  if (tid < 192) v[tid] = 1.0f;
  __syncthreads();
  float sigma = 0.f;
  for (int it = 0; it < 40; ++it) {
    if (tid < 192) {
      float a = 0.f;
      for (int k = 0; k < 192; ++k) a += A[tid][k] * v[k];
      w[tid] = a;
    }
    __syncthreads();
    float vn = 0.f;
    if (tid < 192) {
      float a = 0.f;
      for (int k = 0; k < 192; ++k) a += A[k][tid] * w[k];
      vn = a;
    }
    if (tid == 0) snorm = 0.f;
    __syncthreads();
    if (tid < 192) atomicAdd(&snorm, vn * vn);
    __syncthreads();
    float nrm = sqrtf(snorm);
    sigma = sqrtf(nrm);             // ||A^T A v||/||v|| ~ sigma^2
    if (tid < 192) v[tid] = vn / fmaxf(nrm, 1e-20f);
    __syncthreads();
  }
  float scale = 1.0f / fmaxf(sigma, 1.0f);
  for (int idx = tid; idx < 192 * 192; idx += 256) {
    int i = idx / 192, j = idx - i * 192;
    Kexp[(m * 192 + i) * 192 + j] = A[i][j] * scale;
  }
}

// Kernel 0c: f16 mirror of the scaled experts for WMMA staging (and TDM source).
__global__ __launch_bounds__(256) void k_convK(const float* __restrict__ Kexp,
                                               _Float16* __restrict__ Kh) {
  int idx = blockIdx.x * 256 + threadIdx.x;
  Kh[idx] = (_Float16)Kexp[idx];
}

// ---------------------------------------------------------------------------
// Kernel A: gating network. pi[bt,m] = softmax(gw2 @ gelu(gw1 @ u + gb1) + gb2)
// ---------------------------------------------------------------------------
__global__ __launch_bounds__(256) void k_gate(const float* __restrict__ u,
                                              const float* __restrict__ gw1,
                                              const float* __restrict__ gb1,
                                              const float* __restrict__ gw2,
                                              const float* __restrict__ gb2,
                                              float* __restrict__ pi_out) {
  __shared__ float w1[GHH][DU];
  __shared__ float b1[GHH];
  __shared__ float w2[MM][GHH];
  __shared__ float b2[MM];
  int tid = threadIdx.x;
  for (int i = tid; i < GHH * DU; i += 256) w1[i >> 6][i & 63] = gw1[i];
  for (int i = tid; i < MM * GHH; i += 256) w2[i >> 6][i & 63] = gw2[i];
  if (tid < GHH) b1[tid] = gb1[tid];
  if (tid < MM) b2[tid] = gb2[tid];
  __syncthreads();
  int bt = blockIdx.x * 256 + tid;
  float ur[DU];
#pragma unroll
  for (int n = 0; n < DU; ++n) ur[n] = u[bt * DU + n];
  float lg[MM];
#pragma unroll
  for (int m = 0; m < MM; ++m) lg[m] = b2[m];
  for (int j = 0; j < GHH; ++j) {
    float acc = b1[j];
#pragma unroll
    for (int n = 0; n < DU; ++n) acc += ur[n] * w1[j][n];
    float h = 0.5f * acc * (1.0f + erff(acc * 0.70710678118654752f)); // exact gelu
#pragma unroll
    for (int m = 0; m < MM; ++m) lg[m] += h * w2[m][j];
  }
  float mx = lg[0];
#pragma unroll
  for (int m = 1; m < MM; ++m) mx = fmaxf(mx, lg[m]);
  float e[MM], sum = 0.f;
#pragma unroll
  for (int m = 0; m < MM; ++m) { e[m] = expf(lg[m] - mx); sum += e[m]; }
  float inv = 1.0f / sum;
#pragma unroll
  for (int m = 0; m < MM; ++m) pi_out[bt * MM + m] = e[m] * inv;
}

// ---------------------------------------------------------------------------
// Kernel B: c[bt,a] = sum_m pi[bt,m] * sum_n Kexp[m][a][128+n] * g*u[bt,n]
// A rows scaled by g*pi in f16; Ku tiles staged by the Tensor Data Mover.
// ---------------------------------------------------------------------------
__global__ __launch_bounds__(256) void k_c(const _Float16* __restrict__ Kh,
                                           const float* __restrict__ pi,
                                           const float* __restrict__ u,
                                           const float* __restrict__ lgam,
                                           float* __restrict__ c_out) {
  __shared__ __align__(64) _Float16 Ku[MM][DS][DU];   // [m][a][n], n contiguous
  __shared__ __align__(64) _Float16 us[32][DU];
  __shared__ _Float16 pis[32][MM];                    // g*pi, f16
  int tid = threadIdx.x;
  int bt0 = blockIdx.x * 32;
  float g = expf(lgam[0]);
#if HAVE_TDM
  if (tid < 32) {
    for (int m = 0; m < MM; ++m)   // tile: rows a=0..127, cols n=128..191
      tdm_load_2d_f16(&Ku[m][0][0], Kh + (unsigned)m * 192u * 192u + 128u,
                      /*tile_w=*/DU, /*tile_h=*/DS, /*tensor_w=*/192, /*tensor_h=*/192,
                      /*row_stride=*/192);
    __builtin_amdgcn_s_wait_tensorcnt(0);
  }
#else
  for (int idx = tid; idx < MM * DS * DU; idx += 256) {
    int m = idx >> 13; int r = idx & 8191; int a = r >> 6; int n = r & 63;
    Ku[m][a][n] = Kh[(m * 192 + a) * 192 + DS + n];
  }
#endif
  for (int idx = tid; idx < 32 * DU; idx += 256) us[idx >> 6][idx & 63] = (_Float16)u[bt0 * DU + idx];
  pis[tid >> 3][tid & 7] = (_Float16)(g * pi[bt0 * MM + tid]);
  __syncthreads();
  int wave = tid >> 5, lane = tid & 31, nlane = lane & 15;
  int nt = wave;
  int kbA = (lane & 16) ? 8 : 0, kbB = (lane & 16) ? 16 : 0;
  for (int mt = 0; mt < 2; ++mt) {
    int Arow = mt * 16 + nlane;
    const _Float16* usrc = &us[Arow][0];
    v16h uf[2];
#pragma unroll
    for (int no = 0; no < 2; ++no) uf[no] = a_frag_raw(usrc + no * 32, kbA);
    v8f acc = {};
#pragma unroll
    for (int m = 0; m < MM; ++m) {
      _Float16 s = pis[Arow][m];
#pragma unroll
      for (int no = 0; no < 2; ++no) {
        v16h a = frag_scale(uf[no], s);
        v16h b = *(const v16h*)(&Ku[m][nt * 16 + nlane][no * 32 + kbB]);
        acc = wmma16(a, b, acc);
      }
    }
    int col = nt * 16 + nlane;
    int rowbase = mt * 16 + ((lane & 16) ? 8 : 0);
#pragma unroll
    for (int r = 0; r < 8; ++r)
      c_out[(bt0 + rowbase + r) * DS + col] = acc[r];
  }
}

// ---------------------------------------------------------------------------
// Kernel C (sequential scan): z_{t+1} = (sum_m pi * Kz[m]) z_t + c_t.
// 4 blocks x 16 batches; Kz (8x128x128 f16 = 256KB) TDM-loaded into CDNA5 LDS.
// f32 master state + f16 shadow; per-expert pi-scaling as packed f16 muls.
// Stores ENTERING state z_t (z_0 = 0) for downstream y / x GEMMs.
// ---------------------------------------------------------------------------
__global__ __launch_bounds__(256) void k_scan(const _Float16* __restrict__ Kh,
                                              const float* __restrict__ pi,
                                              const float* __restrict__ c,
                                              float* __restrict__ z_seq) {
  __shared__ __align__(64) _Float16 Kz[MM][DS][DS];   // [m][a][i], i contiguous
  __shared__ float zf[16][DS];                        // f32 master state
  __shared__ __align__(64) _Float16 zh[16][DS];       // f16 shadow for A-frags
  __shared__ _Float16 pis[16][MM];
  int tid = threadIdx.x;
  int b0 = blockIdx.x * 16;
#if HAVE_TDM
  if (tid < 32) {
    for (int m = 0; m < MM; ++m)   // tile: rows a=0..127, cols i=0..127
      tdm_load_2d_f16(&Kz[m][0][0], Kh + (unsigned)m * 192u * 192u,
                      DS, DS, 192, 192, 192);
    __builtin_amdgcn_s_wait_tensorcnt(0);
  }
#else
  for (int idx = tid; idx < MM * DS * DS; idx += 256) {
    int m = idx >> 14; int r = idx & 16383; int a = r >> 7; int i = r & 127;
    Kz[m][a][i] = Kh[(m * 192 + a) * 192 + i];
  }
#endif
  for (int idx = tid; idx < 16 * DS; idx += 256) {
    zf[idx >> 7][idx & 127] = 0.f;
    zh[idx >> 7][idx & 127] = (_Float16)0.f;
  }
  __syncthreads();
  int wave = tid >> 5, lane = tid & 31, nlane = lane & 15;
  int nt = wave;
  int kbA = (lane & 16) ? 8 : 0, kbB = (lane & 16) ? 16 : 0;
  int rowbase = (lane & 16) ? 8 : 0;
  int col = nt * 16 + nlane;
  for (int t = 0; t < TT; ++t) {
    if (tid < 128) pis[tid >> 3][tid & 7] = (_Float16)pi[((b0 + (tid >> 3)) * TT + t) * MM + (tid & 7)];
    for (int idx = tid; idx < 16 * DS; idx += 256) {
      int r = idx >> 7, i = idx & 127;
      z_seq[((b0 + r) * TT + t) * DS + i] = zf[r][i];
    }
    __syncthreads();
    // raw f16 A-fragments of z, one per 32-wide K chunk
    v16h zfr[4];
#pragma unroll
    for (int kc = 0; kc < 4; ++kc) zfr[kc] = a_frag_raw(&zh[nlane][kc * 32], kbA);
    v8f acc = {};
#pragma unroll
    for (int m = 0; m < MM; ++m) {
      _Float16 s = pis[nlane][m];
#pragma unroll
      for (int kc = 0; kc < 4; ++kc) {
        v16h a = frag_scale(zfr[kc], s);
        v16h b = *(const v16h*)(&Kz[m][col][kc * 32 + kbB]);
        acc = wmma16(a, b, acc);
      }
    }
    float znext[8];
#pragma unroll
    for (int r = 0; r < 8; ++r)
      znext[r] = acc[r] + c[((b0 + rowbase + r) * TT + t) * DS + col];
    if (t + 1 < TT)
      __builtin_prefetch(&c[((b0 + rowbase) * TT + (t + 1)) * DS + col], 0, 1);
    __syncthreads();          // all reads of zf/zh done
#pragma unroll
    for (int r = 0; r < 8; ++r) {
      zf[rowbase + r][col] = znext[r];
      zh[rowbase + r][col] = (_Float16)znext[r];
    }
    __syncthreads();          // z_{t+1} visible
  }
}

// ---------------------------------------------------------------------------
// Kernel D: y[bt,j] = sum_m pi * Kexp[m][128+j][:128] . z[bt]  (u-block is 0)
// ---------------------------------------------------------------------------
__global__ __launch_bounds__(256) void k_y(const _Float16* __restrict__ Kh,
                                           const float* __restrict__ pi,
                                           const float* __restrict__ z,
                                           float* __restrict__ y_out) {
  __shared__ __align__(64) _Float16 Ky[MM][DY][DS];   // [m][j][i], i contiguous
  __shared__ __align__(64) _Float16 zs[32][DS];
  __shared__ _Float16 pis[32][MM];
  int tid = threadIdx.x;
  int bt0 = blockIdx.x * 32;
#if HAVE_TDM
  if (tid < 32) {
    for (int m = 0; m < MM; ++m)   // tile: rows j=128..191, cols i=0..127
      tdm_load_2d_f16(&Ky[m][0][0], Kh + ((unsigned)m * 192u + 128u) * 192u,
                      DS, DY, 192, 192, 192);
    __builtin_amdgcn_s_wait_tensorcnt(0);
  }
#else
  for (int idx = tid; idx < MM * DY * DS; idx += 256) {
    int m = idx >> 13; int r = idx & 8191; int j = r >> 7; int i = r & 127;
    Ky[m][j][i] = Kh[(m * 192 + DS + j) * 192 + i];
  }
#endif
  for (int idx = tid; idx < 32 * DS; idx += 256) zs[idx >> 7][idx & 127] = (_Float16)z[bt0 * DS + idx];
  pis[tid >> 3][tid & 7] = (_Float16)pi[bt0 * MM + tid];
  __syncthreads();
  int wave = tid >> 5, lane = tid & 31, nlane = lane & 15;
  int mt = wave >> 2, nt = wave & 3;
  int kbA = (lane & 16) ? 8 : 0, kbB = (lane & 16) ? 16 : 0;
  int Arow = mt * 16 + nlane;
  v16h zfr[4];
#pragma unroll
  for (int kc = 0; kc < 4; ++kc) zfr[kc] = a_frag_raw(&zs[Arow][kc * 32], kbA);
  v8f acc = {};
#pragma unroll
  for (int m = 0; m < MM; ++m) {
    _Float16 s = pis[Arow][m];
#pragma unroll
    for (int kc = 0; kc < 4; ++kc) {
      v16h a = frag_scale(zfr[kc], s);
      v16h b = *(const v16h*)(&Ky[m][nt * 16 + nlane][kc * 32 + kbB]);
      acc = wmma16(a, b, acc);
    }
  }
  int col = nt * 16 + nlane;
  int rb = mt * 16 + ((lane & 16) ? 8 : 0);
#pragma unroll
  for (int r = 0; r < 8; ++r)
    y_out[(bt0 + rb + r) * DY + col] = acc[r];
}

// ---------------------------------------------------------------------------
// Kernel E: state_seq[bt] = Sinv @ z[bt]
// ---------------------------------------------------------------------------
__global__ __launch_bounds__(256) void k_state(const _Float16* __restrict__ Sinv_h,
                                               const float* __restrict__ z,
                                               float* __restrict__ x_out) {
  __shared__ __align__(64) _Float16 Sh[DS][DS];       // Sinv[s][i], i contiguous
  __shared__ __align__(64) _Float16 zs[32][DS];
  int tid = threadIdx.x;
  int bt0 = blockIdx.x * 32;
#if HAVE_TDM
  if (tid < 32) {
    tdm_load_2d_f16(&Sh[0][0], Sinv_h, DS, DS, DS, DS, DS);
    __builtin_amdgcn_s_wait_tensorcnt(0);
  }
#else
  for (int idx = tid; idx < DS * DS; idx += 256) Sh[idx >> 7][idx & 127] = Sinv_h[idx];
#endif
  for (int idx = tid; idx < 32 * DS; idx += 256) zs[idx >> 7][idx & 127] = (_Float16)z[bt0 * DS + idx];
  __syncthreads();
  int wave = tid >> 5, lane = tid & 31, nlane = lane & 15;
  int nt = wave;
  int kbA = (lane & 16) ? 8 : 0, kbB = (lane & 16) ? 16 : 0;
  for (int mt = 0; mt < 2; ++mt) {
    const _Float16* zsrc = &zs[mt * 16 + nlane][0];
    v8f acc = {};
#pragma unroll
    for (int kc = 0; kc < 4; ++kc) {
      v16h a = a_frag_raw(zsrc + kc * 32, kbA);
      v16h b = *(const v16h*)(&Sh[nt * 16 + nlane][kc * 32 + kbB]);
      acc = wmma16(a, b, acc);
    }
    int col = nt * 16 + nlane;
    int rb = mt * 16 + ((lane & 16) ? 8 : 0);
#pragma unroll
    for (int r = 0; r < 8; ++r)
      x_out[(bt0 + rb + r) * DS + col] = acc[r];
  }
}

// ---------------------------------------------------------------------------
// Host launcher
// ---------------------------------------------------------------------------
extern "C" void kernel_launch(void* const* d_in, const int* in_sizes, int n_in,
                              void* d_out, int out_size, void* d_ws, size_t ws_size,
                              hipStream_t stream) {
  (void)in_sizes; (void)n_in; (void)out_size; (void)ws_size;
  const float* u      = (const float*)d_in[0];
  const float* K_raw  = (const float*)d_in[1];
  const float* lgam   = (const float*)d_in[2];
  const float* S_raw  = (const float*)d_in[3];
  const float* gw1    = (const float*)d_in[4];
  const float* gb1    = (const float*)d_in[5];
  const float* gw2    = (const float*)d_in[6];
  const float* gb2    = (const float*)d_in[7];

  float* ws   = (float*)d_ws;
  float* Kexp = ws;                         // 8*192*192       = 294912 f32
  float* Sm   = ws + 294912;                // 128*128         =  16384 f32
  float* Sinv = ws + 311296;                // 128*128         =  16384 f32
  float* piw  = ws + 327680;                // 64*1024*8       = 524288 f32
  float* cw   = ws + 851968;                // 64*1024*128     = 8388608 f32
  float* zw   = ws + 9240576;               // 64*1024*128     = 8388608 f32
  _Float16* Kh    = (_Float16*)(ws + 17629184);   // 294912 f16
  _Float16* Sinvh = (_Float16*)(ws + 17776640);   //  16384 f16

  float* y_out = (float*)d_out;                   // (B,T,DY)
  float* x_out = (float*)d_out + BB * TT * DY;    // (B,T,DS)

  k_buildS <<<1,    128, 0, stream>>>(S_raw, Sm, Sinv, Sinvh);
  k_experts<<<MM,   256, 0, stream>>>(K_raw, Kexp);
  k_convK  <<<MM * 192 * 192 / 256, 256, 0, stream>>>(Kexp, Kh);
  k_gate   <<<BB * TT / 256, 256, 0, stream>>>(u, gw1, gb1, gw2, gb2, piw);
  k_c      <<<BB * TT / 32,  256, 0, stream>>>(Kh, piw, u, lgam, cw);
  k_scan   <<<BB / 16,       256, 0, stream>>>(Kh, piw, cw, zw);
  k_y      <<<BB * TT / 32,  256, 0, stream>>>(Kh, piw, zw, y_out);
  k_state  <<<BB * TT / 32,  256, 0, stream>>>(Sinvh, zw, x_out);
}
